// Attentionscore_57397942943950
// MI455X (gfx1250) — matrix operand order repaired and checked by reading, not compiled, not run on hardware
//
#include <hip/hip_runtime.h>
#include <hip/hip_bf16.h>

typedef __attribute__((ext_vector_type(16))) __bf16 v16bf;
typedef __attribute__((ext_vector_type(8)))  float  v8f;

#define S_LEN   2048
#define D_LEN   64
#define QTILE   128     // queries per block (8 waves x 16 rows)
#define KTILE   64      // keys per iteration
#define KT_STRIDE 68    // bf16 units: 136B rows -> 8B-aligned b64 stores, conflict-free
#define VT_STRIDE 36    // bf16 units: 72B rows  -> 8B-aligned b64 stores, conflict-free
#define P_STRIDE  66    // bf16 units: 132B rows -> odd-dword stride, conflict-free reload
#define LOG2E   1.44269504088896340f

// Fused DPP max butterfly: v_max_num_f32 with the DPP modifier on src0 does
// shuffle+max in ONE VALU op (no v_mov_b32_dpp, no canonicalize max(x,x)).
// Reduces over the 16 lanes of each half-wave (all controls stay within a row of 16).
__device__ __forceinline__ float rowmax16(float x) {
    float r = x;
    asm("v_max_num_f32_dpp %0, %0, %0 quad_perm:[1,0,3,2] row_mask:0xf bank_mask:0xf bound_ctrl:1" : "+v"(r));
    asm("v_max_num_f32_dpp %0, %0, %0 quad_perm:[2,3,0,1] row_mask:0xf bank_mask:0xf bound_ctrl:1" : "+v"(r));
    asm("v_max_num_f32_dpp %0, %0, %0 row_half_mirror row_mask:0xf bank_mask:0xf bound_ctrl:1" : "+v"(r));
    asm("v_max_num_f32_dpp %0, %0, %0 row_mirror row_mask:0xf bank_mask:0xf bound_ctrl:1" : "+v"(r));
    return r;
}

__device__ __forceinline__ void store_bf16x4(__bf16* dst, float a, float b, float c, float d) {
    union { __bf16 h[4]; uint2 u; } pk;
    pk.h[0] = (__bf16)a; pk.h[1] = (__bf16)b; pk.h[2] = (__bf16)c; pk.h[3] = (__bf16)d;
    *reinterpret_cast<uint2*>(dst) = pk.u;   // 8B-aligned by construction
}

__global__ __launch_bounds__(256)
void fa_fwd_kernel(const float* __restrict__ Qg, const float* __restrict__ Kg,
                   const float* __restrict__ Vg, float* __restrict__ Og) {
    __shared__ __bf16 kt[KTILE * KT_STRIDE];          // K tile, row-major [key][d]
    __shared__ __bf16 vt[D_LEN * VT_STRIDE];          // V tile, transposed [d][key]
    __shared__ __bf16 pl[8 * 16 * P_STRIDE];          // per-wave P staging (16x64)

    const int bh    = blockIdx.y;
    const int qbase = blockIdx.x * QTILE;
    const int tid   = threadIdx.x;
    const int wave  = tid >> 5;
    const int lane  = tid & 31;
    const int ln16  = lane & 15;
    const int half  = lane >> 4;

    const size_t head_off = (size_t)bh * S_LEN * D_LEN;
    const float* Qh = Qg + head_off;
    const float* Kh = Kg + head_off;
    const float* Vh = Vg + head_off;
    float*       Oh = Og + head_off;

    const float scale = 0.28867513459481287f;  // 1/sqrt(12)

    // ---- Load Q A-fragments (16x32 bf16 each), pre-scaled ----
    const int qrow = qbase + wave * 16 + ln16;
    v16bf qa[2];
    #pragma unroll
    for (int c = 0; c < 2; ++c) {
        #pragma unroll
        for (int v = 0; v < 8; ++v) {
            const int d = c * 32 + ((v & 4) << 2) + half * 8 + (v & 3) * 2;
            qa[c][2 * v]     = (__bf16)(Qh[(size_t)qrow * D_LEN + d] * scale);
            qa[c][2 * v + 1] = (__bf16)(Qh[(size_t)qrow * D_LEN + d + 1] * scale);
        }
    }

    // All-ones bf16 B fragment: rowsum(P) = P @ ones
    v16bf onesb;
    #pragma unroll
    for (int i = 0; i < 16; ++i) onesb[i] = (__bf16)1.0f;

    // ---- Online-softmax state ----
    float m_r[8];
    v8f acc[4], acc_l;
    const v8f vzero = {};
    #pragma unroll
    for (int r = 0; r < 8; ++r) m_r[r] = -1e30f;
    #pragma unroll
    for (int t = 0; t < 4; ++t) acc[t] = vzero;
    acc_l = vzero;

    __bf16* myp = pl + wave * 16 * P_STRIDE;

    // ---- Staging assignments ----
    // K: thread handles 4 x float4 row-major chunks (key = e>>6, d = e&63)
    // V: thread handles d = tid&63, key-group = tid>>6 (16 keys in 4 strips of 4)
    const int vd  = tid & 63;
    const int vkg = (tid >> 6) << 4;

    float kreg[16], vreg[16];
    #pragma unroll
    for (int i = 0; i < 4; ++i) {
        const int e   = i * 1024 + tid * 4;
        const int key = e >> 6;
        const int d   = e & 63;
        const float4 kv = *reinterpret_cast<const float4*>(&Kh[(size_t)key * D_LEN + d]);
        kreg[4*i] = kv.x; kreg[4*i+1] = kv.y; kreg[4*i+2] = kv.z; kreg[4*i+3] = kv.w;
        #pragma unroll
        for (int c = 0; c < 4; ++c)
            vreg[4*i + c] = Vh[(size_t)(vkg + i * 4 + c) * D_LEN + vd];
    }

    for (int kb = 0; kb < S_LEN; kb += KTILE) {
        __syncthreads();  // previous tile fully consumed
        // ---- Commit prefetched tile to LDS (f32 -> bf16, packed b64 stores) ----
        #pragma unroll
        for (int i = 0; i < 4; ++i) {
            const int e   = i * 1024 + tid * 4;
            const int key = e >> 6;
            const int d   = e & 63;
            store_bf16x4(&kt[key * KT_STRIDE + d], kreg[4*i], kreg[4*i+1], kreg[4*i+2], kreg[4*i+3]);
            store_bf16x4(&vt[vd * VT_STRIDE + vkg + i * 4], vreg[4*i], vreg[4*i+1], vreg[4*i+2], vreg[4*i+3]);
        }
        __syncthreads();

        // ---- Prefetch next tile while this tile computes ----
        if (kb + KTILE < S_LEN) {
            const int nb = kb + KTILE;
            #pragma unroll
            for (int i = 0; i < 4; ++i) {
                const int e   = i * 1024 + tid * 4;
                const int key = nb + (e >> 6);
                const int d   = e & 63;
                const float4 kv = *reinterpret_cast<const float4*>(&Kh[(size_t)key * D_LEN + d]);
                kreg[4*i] = kv.x; kreg[4*i+1] = kv.y; kreg[4*i+2] = kv.z; kreg[4*i+3] = kv.w;
                #pragma unroll
                for (int c = 0; c < 4; ++c)
                    vreg[4*i + c] = Vh[(size_t)(nb + vkg + i * 4 + c) * D_LEN + vd];
            }
        }

        // ---- Scores: S[16q x 64k] = Q(16x64) @ K^T(64x64), four 16-key groups ----
        v8f s[4] = {vzero, vzero, vzero, vzero};
        #pragma unroll
        for (int c = 0; c < 2; ++c) {
            #pragma unroll
            for (int g = 0; g < 4; ++g) {
                v16bf b;
                #pragma unroll
                for (int v = 0; v < 8; ++v) {
                    const int d = c * 32 + ((v & 4) << 2) + half * 8 + (v & 3) * 2;
                    b[2 * v]     = kt[(g * 16 + ln16) * KT_STRIDE + d];
                    b[2 * v + 1] = kt[(g * 16 + ln16) * KT_STRIDE + d + 1];
                }
                s[g] = __builtin_amdgcn_wmma_f32_16x16x32_bf16(false, qa[c], false, b,
                                                               (short)0, s[g], false, false);
            }
        }

        // ---- Online softmax: one DPP butterfly + one rescale per 64 keys ----
        #pragma unroll
        for (int r = 0; r < 8; ++r) {
            const float mloc = fmaxf(fmaxf(s[0][r], s[1][r]), fmaxf(s[2][r], s[3][r]));
            const float mx   = rowmax16(mloc);
            const float mn   = fmaxf(m_r[r], mx);
            const float corr = __builtin_exp2f((m_r[r] - mn) * LOG2E);
            m_r[r] = mn;

            #pragma unroll
            for (int g = 0; g < 4; ++g)
                s[g][r] = __builtin_exp2f((s[g][r] - mn) * LOG2E);

            acc_l[r] *= corr;
            #pragma unroll
            for (int t = 0; t < 4; ++t) acc[t][r] *= corr;
        }

        // ---- P: C-layout -> A-layout via per-wave LDS staging ----
        #pragma unroll
        for (int r = 0; r < 8; ++r) {
            const int row = r + half * 8;
            #pragma unroll
            for (int g = 0; g < 4; ++g)
                myp[row * P_STRIDE + g * 16 + ln16] = (__bf16)s[g][r];
        }
        asm volatile("s_wait_dscnt 0" ::: "memory");
        v16bf pa[2];
        #pragma unroll
        for (int h2 = 0; h2 < 2; ++h2) {
            #pragma unroll
            for (int v = 0; v < 8; ++v) {
                const int k = h2 * 32 + ((v & 4) << 2) + half * 8 + (v & 3) * 2;
                pa[h2][2 * v]     = myp[ln16 * P_STRIDE + k];
                pa[h2][2 * v + 1] = myp[ln16 * P_STRIDE + k + 1];
            }
        }

        // ---- Row sums of P into acc_l ----
        acc_l = __builtin_amdgcn_wmma_f32_16x16x32_bf16(false, pa[0], false, onesb,
                                                        (short)0, acc_l, false, false);
        acc_l = __builtin_amdgcn_wmma_f32_16x16x32_bf16(false, pa[1], false, onesb,
                                                        (short)0, acc_l, false, false);

        // ---- O += P(16x64) @ V(64x64): 4 d-tiles, two 32-key halves each ----
        #pragma unroll
        for (int t = 0; t < 4; ++t) {
            const int d = t * 16 + ln16;
            v16bf vb0, vb1;
            #pragma unroll
            for (int v = 0; v < 8; ++v) {
                const int key = ((v & 4) << 2) + half * 8 + (v & 3) * 2;
                vb0[2 * v]     = vt[d * VT_STRIDE + key];
                vb0[2 * v + 1] = vt[d * VT_STRIDE + key + 1];
                vb1[2 * v]     = vt[d * VT_STRIDE + 32 + key];
                vb1[2 * v + 1] = vt[d * VT_STRIDE + 32 + key + 1];
            }
            acc[t] = __builtin_amdgcn_wmma_f32_16x16x32_bf16(false, pa[0], false, vb0,
                                                             (short)0, acc[t], false, false);
            acc[t] = __builtin_amdgcn_wmma_f32_16x16x32_bf16(false, pa[1], false, vb1,
                                                             (short)0, acc[t], false, false);
        }
    }

    // ---- Normalize and store O (f32) ----
    #pragma unroll
    for (int t = 0; t < 4; ++t) {
        #pragma unroll
        for (int r = 0; r < 8; ++r) {
            const int row = qbase + wave * 16 + r + half * 8;
            const int d   = t * 16 + ln16;
            Oh[(size_t)row * D_LEN + d] = acc[t][r] / acc_l[r];
        }
    }
}

extern "C" void kernel_launch(void* const* d_in, const int* in_sizes, int n_in,
                              void* d_out, int out_size, void* d_ws, size_t ws_size,
                              hipStream_t stream) {
    (void)n_in; (void)out_size; (void)d_ws; (void)ws_size;
    const float* Q = (const float*)d_in[0];
    const float* K = (const float*)d_in[1];
    const float* V = (const float*)d_in[2];
    float* O = (float*)d_out;

    const int BH = in_sizes[0] / (S_LEN * D_LEN);  // 4*16 = 64 heads total
    dim3 grid(S_LEN / QTILE, BH);
    dim3 block(256);
    fa_fwd_kernel<<<grid, block, 0, stream>>>(Q, K, V, O);
}
